// Bi_Dir_FeatureExtractor_74603581931746
// MI455X (gfx1250) — matrix operand order repaired and checked by reading, not compiled, not run on hardware
//
#include <hip/hip_runtime.h>
#include <hip/hip_bf16.h>

// ---------------------------------------------------------------------------
// CDNA5 (gfx1250) bidirectional feature extractor.
//  - 3x3 convs -> implicit GEMM on v_wmma_f32_16x16x32_bf16 (fp32 accum)
//  - weights pre-converted to bf16; tiles staged via TDM tensor_load_to_lds
//    (double-buffered LDS, s_wait_tensorcnt); A-frags = 2x ds_load_b128
//    straight into the WMMA operand (no cvt); input gathered branch-free
//  - softsplat -> global_atomic_add_f32
// ---------------------------------------------------------------------------

typedef __attribute__((ext_vector_type(16))) __bf16 v16bf;
typedef __attribute__((ext_vector_type(8)))  __bf16 v8bf;
typedef __attribute__((ext_vector_type(8)))  float  v8f;
typedef unsigned int u32x4 __attribute__((ext_vector_type(4)));
typedef int          i32x8 __attribute__((ext_vector_type(8)));
typedef int          i32x4 __attribute__((ext_vector_type(4)));

#define BATCH 8

// ---------------------------------------------------------------------------
// TDM: load a (tileM x 32) bf16 tile of the weight matrix into LDS.
// Tensor described from the tile origin: remK x remM elements, row stride K.
// LDS padding: 4 dwords after every 16 dwords -> 20-dword (40 bf16) row pitch
// (rows 16B aligned for ds_load_b128; 20*r mod 64 -> 16 disjoint bank groups).
// ---------------------------------------------------------------------------
__device__ __forceinline__ void tdm_load_wtile(const __bf16* gptr, unsigned lds_byte_off,
                                               int remK, int remM, int strideK, int tileM)
{
    unsigned long long ga = (unsigned long long)(const void*)gptr;
    u32x4 g0;
    g0[0] = 1u;                                            // count=1, user descriptor
    g0[1] = lds_byte_off;                                  // lds_addr (bytes)
    g0[2] = (unsigned)(ga & 0xFFFFFFFFull);                // global_addr[31:0]
    g0[3] = (unsigned)((ga >> 32) & 0x01FFFFFFull)         // global_addr[56:32]
          | (2u << 30);                                    // type=2 (image)
    i32x8 g1;
    g1[0] = (int)((1u << 16)                               // data_size: 2 bytes
          |       (1u << 20)                               // pad_enable
          |       (3u << 22)                               // pad_interval: 16 dwords
          |       (3u << 25));                             // pad_amount: 4 dwords
    g1[1] = (int)(((unsigned)remK & 0xFFFFu) << 16);       // tensor_dim0[15:0] @ [63:48]
    g1[2] = (int)(((unsigned)remK >> 16)                   // tensor_dim0[31:16]
          |       (((unsigned)remM & 0xFFFFu) << 16));     // tensor_dim1[15:0]
    g1[3] = (int)(((unsigned)remM >> 16)                   // tensor_dim1[31:16]
          |       (32u << 16));                            // tile_dim0 = 32 elements
    g1[4] = tileM;                                         // tile_dim1
    g1[5] = strideK;                                       // tensor_dim0_stride[31:0]
    g1[6] = 0;
    g1[7] = 0;
    i32x4 z4 = {0, 0, 0, 0};
#if __has_include(<hip/amd_detail/amd_gfx1250_TDM.h>)
    i32x8 z8 = {0, 0, 0, 0, 0, 0, 0, 0};
    __builtin_amdgcn_tensor_load_to_lds(g0, g1, z4, z4, z8, 0);
#else
    __builtin_amdgcn_tensor_load_to_lds(g0, g1, z4, z4, 0);
#endif
}

// ---------------------------------------------------------------------------
// fp32 -> bf16 weight conversion (run once per conv layer, stream-ordered).
// ---------------------------------------------------------------------------
__global__ void cvt_w_bf16_kernel(const float* __restrict__ w,
                                  __bf16* __restrict__ o, int n)
{
    int i = blockIdx.x * blockDim.x + threadIdx.x;
    int i4 = i * 4;
    if (i4 + 3 < n) {
        float4 v = *(const float4*)(w + i4);
        o[i4 + 0] = (__bf16)v.x;
        o[i4 + 1] = (__bf16)v.y;
        o[i4 + 2] = (__bf16)v.z;
        o[i4 + 3] = (__bf16)v.w;
    }
}

// ---------------------------------------------------------------------------
// Implicit-GEMM 3x3 conv, pad=1, stride 1 or 2, optional fused SiLU.
// O[oc][pix] = sum_k W[oc][k] * X[k][pix], k = ci*9 + kh*3 + kw.
// Block: 128 threads = 4 waves. Each wave: (MT*16 oc) x (16 pixel) tile,
// MT WMMAs per K-chunk sharing one gathered B fragment.
// ---------------------------------------------------------------------------
template <int MT>
__global__ __launch_bounds__(128, 2)
void conv3x3_wmma_kernel(const float* __restrict__ in, long in_bstride,
                         const __bf16* __restrict__ wgt,
                         const float* __restrict__ bias,
                         float* __restrict__ out,
                         int Cin, int Hin, int Win,
                         int Cout, int Hout, int Wout,
                         int stride, int do_silu)
{
    extern __shared__ __bf16 smem[];           // [2][MT*16][40] bf16 weight tiles
    constexpr int PITCH = 40;                  // bf16 elems per LDS row (32 + 8 pad)
    constexpr int AT = MT * 16 * PITCH;        // bf16 elems per buffer

    const int tid  = threadIdx.x;
    const int wave = tid >> 5;
    const int lane = tid & 31;
    const int mn   = lane & 15;                // M row (A) / N col (B,D)
    const int half = lane >> 4;
    const int oc0  = blockIdx.y * (16 * MT);
    const int P    = Hout * Wout;
    const int p0   = blockIdx.x * 64 + wave * 16;
    const int b    = blockIdx.z;
    const int K    = Cin * 9;
    const int HWin = Hin * Win;                // per-channel plane (fits 32-bit)

    const float* inB = in + (long)b * in_bstride;
    const int pix = p0 + mn;
    const int oy  = pix / Wout;
    const int ox  = pix - oy * Wout;

    // Precompute clamped row/col offsets + validity for the 3x3 window.
    int ro0, ro1, ro2, co0, co1, co2;
    bool vy0, vy1, vy2, vx0, vx1, vx2;
    {
        int iy0 = oy * stride - 1, iy1 = iy0 + 1, iy2 = iy0 + 2;
        int ix0 = ox * stride - 1, ix1 = ix0 + 1, ix2 = ix0 + 2;
        vy0 = (iy0 >= 0) & (iy0 < Hin); vy1 = (iy1 >= 0) & (iy1 < Hin); vy2 = (iy2 >= 0) & (iy2 < Hin);
        vx0 = (ix0 >= 0) & (ix0 < Win); vx1 = (ix1 >= 0) & (ix1 < Win); vx2 = (ix2 >= 0) & (ix2 < Win);
        ro0 = min(max(iy0, 0), Hin - 1) * Win;
        ro1 = min(max(iy1, 0), Hin - 1) * Win;
        ro2 = min(max(iy2, 0), Hin - 1) * Win;
        co0 = min(max(ix0, 0), Win - 1);
        co1 = min(max(ix1, 0), Win - 1);
        co2 = min(max(ix2, 0), Win - 1);
    }

    v8f acc[MT];
#pragma unroll
    for (int mt = 0; mt < MT; ++mt) acc[mt] = (v8f){};

    const int nchunks = (K + 31) >> 5;

    auto issue = [&](int c) {
        int k0 = c * 32;
        tdm_load_wtile(wgt + (long)oc0 * K + k0, (unsigned)((c & 1) * AT * 2),
                       K - k0, Cout - oc0, K, MT * 16);
    };

    if (tid < 32) issue(0);

    for (int c = 0; c < nchunks; ++c) {
        __syncthreads();                        // prev reads of buf[(c+1)&1] done
        const bool more = (c + 1 < nchunks);
        if (tid < 32) {
            if (more) {
                issue(c + 1);
                __builtin_amdgcn_s_wait_tensorcnt(1);   // chunk c resident
            } else {
                __builtin_amdgcn_s_wait_tensorcnt(0);
            }
        }
        __syncthreads();

        const int k0 = c * 32;
        const __bf16* As = smem + (c & 1) * AT;

        // --- B fragment: branch-free gather, 32-bit offsets, cndmask selects
        // 16-bit B 32x16 layout: lanes 0-15 K=0..7,16..23; lanes 16-31 K=8..15,24..31
        float bflat[16];
#pragma unroll
        for (int e = 0; e < 16; ++e) {
            int k  = k0 + half * 8 + e + ((e >= 8) ? 8 : 0);
            int kk = min(k, K - 1);
            int ci = kk / 9;
            int r  = kk - ci * 9;
            int kh = r / 3;
            int kw = r - kh * 3;
            int ro = (kh == 0) ? ro0 : ((kh == 1) ? ro1 : ro2);
            int co = (kw == 0) ? co0 : ((kw == 1) ? co1 : co2);
            bool vy = (kh == 0) ? vy0 : ((kh == 1) ? vy1 : vy2);
            bool vx = (kw == 0) ? vx0 : ((kw == 1) ? vx1 : vx2);
            bool ok = (k < K) & vy & vx;
            float val = inB[ci * HWin + ro + co];
            bflat[e] = ok ? val : 0.0f;
        }
        v16bf bfrag;
#pragma unroll
        for (int e = 0; e < 16; ++e) bfrag[e] = (__bf16)bflat[e];

        // --- MT A fragments: 2x ds_load_b128 each, no conversion ---
        // 16-bit A 16x32 layout: lanes0-15 V0..3->K0..7,V4..7->K16..23;
        //                        lanes16-31 V0..3->K8..15,V4..7->K24..31
#pragma unroll
        for (int mt = 0; mt < MT; ++mt) {
            const __bf16* Ar = As + (mt * 16 + mn) * PITCH + half * 8;
            v8bf lo = *(const v8bf*)(Ar);        // rel K = half*8 .. +7
            v8bf hi = *(const v8bf*)(Ar + 16);   // rel K = 16+half*8 .. +7
            v16bf afrag = __builtin_shufflevector(lo, hi,
                0, 1, 2, 3, 4, 5, 6, 7, 8, 9, 10, 11, 12, 13, 14, 15);
            acc[mt] = __builtin_amdgcn_wmma_f32_16x16x32_bf16(
                false, afrag, false, bfrag, (short)0, acc[mt], false, false);
        }
    }

    // D layout: lane n = lane&15; VGPR v -> M = v + 8*(lane>>4)
    float* outB = out + (long)b * Cout * P;
#pragma unroll
    for (int mt = 0; mt < MT; ++mt) {
#pragma unroll
        for (int v = 0; v < 8; ++v) {
            int m = oc0 + mt * 16 + v + half * 8;
            float y = acc[mt][v] + bias[m];
            if (do_silu) y = y / (1.0f + __expf(-y));
            outB[(long)m * P + pix] = y;
        }
    }
}

// ---------------------------------------------------------------------------
// Bilinear resize of a 2-channel 512x512 flow field to res x res + normalize
// by (res-1)/2.
// ---------------------------------------------------------------------------
__global__ void resize_norm_flow_kernel(const float* __restrict__ flow,
                                        long bstride, long cstride,
                                        float* __restrict__ outf, int res)
{
    int idx = blockIdx.x * blockDim.x + threadIdx.x;
    int rr = res * res;
    int total = BATCH * 2 * rr;
    if (idx >= total) return;
    int p = idx % rr;
    int c = (idx / rr) & 1;
    int b = idx / (2 * rr);
    int oyi = p / res, oxi = p % res;

    float scale = 512.0f / (float)res;
    float sy = ((float)oyi + 0.5f) * scale - 0.5f;
    float sx = ((float)oxi + 0.5f) * scale - 0.5f;
    float syf = floorf(sy), sxf = floorf(sx);
    int y0 = (int)syf, x0 = (int)sxf;
    float wy = sy - syf, wx = sx - sxf;
    int y0c = min(max(y0, 0), 511), y1c = min(max(y0 + 1, 0), 511);
    int x0c = min(max(x0, 0), 511), x1c = min(max(x0 + 1, 0), 511);

    const float* f = flow + (long)b * bstride + (long)c * cstride;
    float v00 = f[y0c * 512 + x0c];
    float v01 = f[y0c * 512 + x1c];
    float v10 = f[y1c * 512 + x0c];
    float v11 = f[y1c * 512 + x1c];
    float v = v00 * (1.0f - wy) * (1.0f - wx) + v01 * (1.0f - wy) * wx +
              v10 * wy * (1.0f - wx) + v11 * wy * wx;

    float denom = (float)(res - 1) * 0.5f;
    outf[((long)b * 2 + c) * rr + p] = v / denom;
}

// ---------------------------------------------------------------------------
// Softsplat scatter: bilinear scatter-add of [feat*e, e] into [B][C+1][r][r].
// ---------------------------------------------------------------------------
__global__ void softsplat_scatter_kernel(const float* __restrict__ feat,
                                         const float* __restrict__ flown,
                                         float* __restrict__ accum,
                                         int C, int res)
{
    int idx = blockIdx.x * blockDim.x + threadIdx.x;
    int rr = res * res;
    int total = BATCH * rr;
    if (idx >= total) return;
    int p = idx % rr;
    int b = idx / rr;
    int y = p / res, x = p % res;

    const float* fl = flown + (long)b * 2 * rr;
    float fx = (float)x + fl[p];
    float fy = (float)y + fl[rr + p];
    float x0f = floorf(fx), y0f = floorf(fy);
    int x0 = (int)x0f, y0 = (int)y0f;
    float wx1 = fx - x0f, wy1 = fy - y0f;
    float w00 = (1.0f - wx1) * (1.0f - wy1);
    float w01 = wx1 * (1.0f - wy1);
    float w10 = (1.0f - wx1) * wy1;
    float w11 = wx1 * wy1;

    bool v00 = (x0 >= 0) & (x0 < res) & (y0 >= 0) & (y0 < res);
    bool v01 = (x0 + 1 >= 0) & (x0 + 1 < res) & (y0 >= 0) & (y0 < res);
    bool v10 = (x0 >= 0) & (x0 < res) & (y0 + 1 >= 0) & (y0 + 1 < res);
    bool v11 = (x0 + 1 >= 0) & (x0 + 1 < res) & (y0 + 1 >= 0) & (y0 + 1 < res);
    int i00 = y0 * res + x0;
    int i01 = y0 * res + x0 + 1;
    int i10 = (y0 + 1) * res + x0;
    int i11 = (y0 + 1) * res + x0 + 1;

    const float E = 2.71828182845904523536f;
    const float* fb = feat + (long)b * C * rr;
    float* ab = accum + (long)b * (C + 1) * rr;

    for (int c = 0; c <= C; ++c) {
        float val = (c < C) ? fb[(long)c * rr + p] * E : E;
        float* ac = ab + (long)c * rr;
        if (v00) atomicAdd(ac + i00, val * w00);
        if (v01) atomicAdd(ac + i01, val * w01);
        if (v10) atomicAdd(ac + i10, val * w10);
        if (v11) atomicAdd(ac + i11, val * w11);
    }
}

// ---------------------------------------------------------------------------
// Divide by accumulated metric (den==0 -> 1) and concat first/last features.
// ---------------------------------------------------------------------------
__global__ void norm_concat_kernel(const float* __restrict__ accF,
                                   const float* __restrict__ accL,
                                   float* __restrict__ outc,
                                   int C, int res)
{
    int idx = blockIdx.x * blockDim.x + threadIdx.x;
    int rr = res * res;
    int total = BATCH * 2 * C * rr;
    if (idx >= total) return;
    int p = idx % rr;
    int c = (idx / rr) % (2 * C);
    int b = idx / (2 * C * rr);

    const float* src = (c < C) ? accF : accL;
    int cc = (c < C) ? c : c - C;
    const float* sb = src + (long)b * (C + 1) * rr;
    float den = sb[(long)C * rr + p];
    den = (den == 0.0f) ? 1.0f : den;
    outc[((long)b * 2 * C + c) * rr + p] = sb[(long)cc * rr + p] / den;
}

// ---------------------------------------------------------------------------
// Host orchestration
// ---------------------------------------------------------------------------
static void launch_conv(const float* in, long in_bstride,
                        const float* w, const float* bias, float* out,
                        __bf16* wbf,  // bf16 weight staging (stream-ordered reuse)
                        int Cin, int Hin, int Win, int Cout,
                        int stride, int silu, hipStream_t s)
{
    int K = Cin * 9;
    int nW = Cout * K;
    cvt_w_bf16_kernel<<<(nW / 4 + 255) / 256, 256, 0, s>>>(w, wbf, nW);

    int Hout = (stride == 2) ? Hin / 2 : Hin;
    int Wout = (stride == 2) ? Win / 2 : Win;
    int P = Hout * Wout;
    int MT = (Cout % 64 == 0) ? 4 : ((Cout % 32 == 0) ? 2 : 1);
    dim3 grid(P / 64, Cout / (16 * MT), BATCH);
    size_t shmem = (size_t)2 * MT * 16 * 40 * sizeof(__bf16);
    if (MT == 4)
        conv3x3_wmma_kernel<4><<<grid, 128, shmem, s>>>(in, in_bstride, wbf, bias, out,
            Cin, Hin, Win, Cout, Hout, Wout, stride, silu);
    else if (MT == 2)
        conv3x3_wmma_kernel<2><<<grid, 128, shmem, s>>>(in, in_bstride, wbf, bias, out,
            Cin, Hin, Win, Cout, Hout, Wout, stride, silu);
    else
        conv3x3_wmma_kernel<1><<<grid, 128, shmem, s>>>(in, in_bstride, wbf, bias, out,
            Cin, Hin, Win, Cout, Hout, Wout, stride, silu);
}

extern "C" void kernel_launch(void* const* d_in, const int* in_sizes, int n_in,
                              void* d_out, int out_size, void* d_ws, size_t ws_size,
                              hipStream_t stream)
{
    const float* lc = (const float*)d_in[0];   // (8,6,512,512)
    const float* fl = (const float*)d_in[1];   // (8,4,512,512)
    auto W = [&](int i) { return (const float*)d_in[i]; };

    float* ws   = (float*)d_ws;
    float* bufA = ws;                  // 33,554,432 floats
    float* bufB = ws + 33554432L;      // 33,554,432 floats
    float* bufC = ws + 67108864L;      // 16,777,216 floats
    __bf16* wbf = (__bf16*)(ws + 83886080L);   // 2,359,296 bf16 (weight staging)

    const long HW = 512L * 512L;

    // ---- pre_extract(first_frame = lc[:,3:6]) : ping-pong bufA/bufB ----
    launch_conv(lc + 3 * HW, 6 * HW,   W(2),  W(3),  bufA, wbf,  3, 512, 512, 16, 1, 1, stream);
    launch_conv(bufA, 16 * HW,         W(4),  W(5),  bufB, wbf, 16, 512, 512, 32, 2, 1, stream);
    launch_conv(bufB, 32L * 256 * 256, W(6),  W(7),  bufA, wbf, 32, 256, 256, 32, 1, 1, stream);
    launch_conv(bufA, 32L * 256 * 256, W(8),  W(9),  bufB, wbf, 32, 256, 256, 64, 2, 1, stream);
    launch_conv(bufB, 64L * 128 * 128, W(10), W(11), bufA, wbf, 64, 128, 128, 64, 1, 1, stream);
    // ff = bufA (8 x 64 x 128 x 128)

    // ---- pre_extract(last_frame = lc[:,0:3]) : ping-pong bufB/bufC ----
    launch_conv(lc, 6 * HW,            W(12), W(13), bufB, wbf,  3, 512, 512, 16, 1, 1, stream);
    launch_conv(bufB, 16 * HW,         W(14), W(15), bufC, wbf, 16, 512, 512, 32, 2, 1, stream);
    launch_conv(bufC, 32L * 256 * 256, W(16), W(17), bufB, wbf, 32, 256, 256, 32, 1, 1, stream);
    launch_conv(bufB, 32L * 256 * 256, W(18), W(19), bufC, wbf, 32, 256, 256, 64, 2, 1, stream);
    launch_conv(bufC, 64L * 128 * 128, W(20), W(21), bufB, wbf, 64, 128, 128, 64, 1, 1, stream);
    // lf = bufB (8 x 64 x 128 x 128)

    // per-level feature buffers carved past ff/lf (which occupy 8,388,608 floats)
    float* ffL[4] = { bufA + 16777216L, bufA + 19922944L, bufA + 20971520L, bufA + 21364736L };
    float* lfL[4] = { bufB + 16777216L, bufB + 19922944L, bufB + 20971520L, bufB + 21364736L };
    // per-level scratch in bufC (free after lf pre-extract)
    float* accF = bufC;                // up to 3,178,496 floats
    float* accL = bufC + 3178496L;     // up to 3,178,496 floats
    float* conc = bufC + 6356992L;     // up to 6,291,456 floats
    float* flF  = bufC + 12648448L;    // up to 65,536 floats
    float* flB  = bufC + 12713984L;    // up to 65,536 floats

    const int halfc[4] = { 96, 128, 192, 256 };
    const int resv[4]  = { 64, 32, 16, 8 };
    const int ewf[4]   = { 22, 24, 26, 28 };
    const int ewl[4]   = { 30, 32, 34, 36 };
    const int zw[4]    = { 38, 40, 42, 44 };
    const long outOff[4] = { 0L, 6291456L, 8388608L, 9175040L };

    const float* ffprev = bufA;
    const float* lfprev = bufB;
    int Cprev = 64, Hprev = 128;

    for (int i = 0; i < 4; ++i) {
        int Co = halfc[i];
        int r  = resv[i];
        long rr = (long)r * r;

        // strided extractor convs (SiLU)
        launch_conv(ffprev, (long)Cprev * Hprev * Hprev, W(ewf[i]), W(ewf[i] + 1),
                    ffL[i], wbf, Cprev, Hprev, Hprev, Co, 2, 1, stream);
        launch_conv(lfprev, (long)Cprev * Hprev * Hprev, W(ewl[i]), W(ewl[i] + 1),
                    lfL[i], wbf, Cprev, Hprev, Hprev, Co, 2, 1, stream);

        // resize + normalize flows
        {
            int tot = BATCH * 2 * (int)rr;
            int blocks = (tot + 255) / 256;
            resize_norm_flow_kernel<<<blocks, 256, 0, stream>>>(
                fl, 4 * HW, HW, flF, r);                    // forward (ch 0:2)
            resize_norm_flow_kernel<<<blocks, 256, 0, stream>>>(
                fl + 2 * HW, 4 * HW, HW, flB, r);           // backward (ch 2:4)
        }

        // zero accumulators, scatter-splat both directions
        size_t accBytes = (size_t)BATCH * (Co + 1) * rr * sizeof(float);
        hipMemsetAsync(accF, 0, accBytes, stream);
        hipMemsetAsync(accL, 0, accBytes, stream);
        {
            int tot = BATCH * (int)rr;
            int blocks = (tot + 255) / 256;
            softsplat_scatter_kernel<<<blocks, 256, 0, stream>>>(ffL[i], flF, accF, Co, r);
            softsplat_scatter_kernel<<<blocks, 256, 0, stream>>>(lfL[i], flB, accL, Co, r);
        }

        // normalize + concat
        {
            int tot = BATCH * 2 * Co * (int)rr;
            int blocks = (tot + 255) / 256;
            norm_concat_kernel<<<blocks, 256, 0, stream>>>(accF, accL, conc, Co, r);
        }

        // zero conv (3x3, stride 1, no activation) -> output slice
        launch_conv(conc, (long)2 * Co * rr, W(zw[i]), W(zw[i] + 1),
                    (float*)d_out + outOff[i], wbf, 2 * Co, r, r, 2 * Co, 1, 0, stream);

        ffprev = ffL[i];
        lfprev = lfL[i];
        Cprev = Co;
        Hprev = r;
    }

    (void)in_sizes; (void)n_in; (void)out_size; (void)ws_size;
}
